// Attention_46231027974287
// MI455X (gfx1250) — compile-verified, matrix operand order checked by reading, not run on hardware
//
#include <hip/hip_runtime.h>

// ---------------- problem constants ----------------
#define DIMD   512
#define NHEADS 8
#define HDIM   64
#define BATCH  2
#define SEQ    4096
#define MROWS  (BATCH * SEQ)   // 8192 rows of [rows, 512] activations

typedef __attribute__((ext_vector_type(16))) _Float16 v16h;
typedef __attribute__((ext_vector_type(8)))  _Float16 v8h;
typedef __attribute__((ext_vector_type(4)))  _Float16 v4h;
typedef __attribute__((ext_vector_type(8)))  float    v8f;
typedef __attribute__((ext_vector_type(4)))  unsigned int v4u;
typedef __attribute__((ext_vector_type(8)))  int      v8i;
typedef __attribute__((ext_vector_type(4)))  int      v4i;

#ifndef __has_builtin
#define __has_builtin(x) 0
#endif
#if __has_builtin(__builtin_amdgcn_tensor_load_to_lds)
#  define HAVE_TDM 1
#else
#  define HAVE_TDM 0
#endif
#if __has_builtin(__builtin_amdgcn_s_wait_tensorcnt)
#  define WAIT_TCNT(n) __builtin_amdgcn_s_wait_tensorcnt(n)
#else
#  define WAIT_TCNT(n) asm volatile("s_wait_tensorcnt %0" :: "n"(n) : "memory")
#endif

// A/B fragment loader for v_wmma_f32_16x16x32_f16 from a row-major
// [16 rows][>=32 cols] f16 tile. Per the CDNA5 ISA 16-bit A layout:
// lane = g*16 + row; element e<8 -> K = g*8+e; e>=8 -> K = 16+g*8+(e-8).
// That is two contiguous 8-half (16B) loads at col offsets g*8 and 16+g*8.
__device__ __forceinline__ v16h lds_frag(const _Float16* tile, int strideH, int lane) {
    const int row = lane & 15;
    const int g   = (lane >> 4) & 1;
    const _Float16* p = tile + row * strideH + g * 8;
    union { v16h v; v8h h[2]; } u;
    u.h[0] = *(const v8h*)(p);
    u.h[1] = *(const v8h*)(p + 16);
    return u.v;
}

__device__ __forceinline__ v8f wmma16(v16h a, v16h b, v8f c) {
    return __builtin_amdgcn_wmma_f32_16x16x32_f16(false, a, false, b, (short)0, c, false, false);
}

#if HAVE_TDM
// TDM 2D tile load: 64x64 f16 tile, row stride `rowStrideElems` (f16 units) in
// global memory, destination LDS rows padded 128B data + 16B pad (=144B => the
// [64][72] f16 tile layout). D# packing per CDNA5 ISA ch.8 (groups 0/1).
// This toolchain's builtin is the 6-arg form:
//   (uint32x4 g0, int32x8 g1, int32x4, int32x4, int32x8, i32 cpol)
__device__ __forceinline__ void tdm_load_2d(unsigned lds_off, const _Float16* g,
                                            unsigned long long rowStrideElems)
{
    const unsigned long long ga = (unsigned long long)(size_t)g;
    v4u g0;
    g0[0] = 1u;                                   // count=1 (valid user D#)
    g0[1] = lds_off;                              // LDS byte address
    g0[2] = (unsigned)ga;                         // global_addr[31:0]
    g0[3] = (unsigned)((ga >> 32) & 0x1ffffffull) // global_addr[56:32]
          | (2u << 30);                           // type=2 ("image")
    v8i g1;
    g1[0] = (1 << 16)                             // data_size = 1 -> 2 bytes
          | (1 << 20)                             // pad_enable
          | (4 << 22)                             // pad_interval: 32 DW = 128B
          | (3 << 25);                            // pad_amount:   4 DW = 16B
    g1[1] = 0;                                    // barrier addr 0; tdim0 lo16=0
    g1[2] = 16;                                   // tensor_dim0 = 1<<20 (hi16), tdim1 lo16=0
    g1[3] = 16 | (64 << 16);                      // tensor_dim1 = 1<<20; tile_dim0 = 64
    g1[4] = 64;                                   // tile_dim1 = 64; tile_dim2 = 0
    g1[5] = (int)(rowStrideElems & 0xffffffffull);// tensor_dim0_stride lo32
    g1[6] = (int)((rowStrideElems >> 32) & 0xffffull);
    g1[7] = 0;
    const v4i gz4 = {0, 0, 0, 0};
    const v8i gz8 = {0, 0, 0, 0, 0, 0, 0, 0};
    __builtin_amdgcn_tensor_load_to_lds(g0, g1, gz4, gz4, gz8, 0);
}
#endif

#define PK_STRIDE 48   // 32 K-cols padded to 48 halves = 96B (16B multiple)

// ---------------------------------------------------------------------------
// QKV projection: out[m][n] = (X[m,:] . W[n,:] + bias[n]) * scale
// vTransposed==0 : write f16 to [B,H,T,Hd]   (Q and K)
// vTransposed==1 : write f16 to [B,H,Hd,T]   (V, pre-transposed for PV wmma)
// ---------------------------------------------------------------------------
__global__ void __launch_bounds__(128)
qkv_proj_kernel(const float* __restrict__ X, const float* __restrict__ W,
                const float* __restrict__ bias, _Float16* __restrict__ out,
                int vTransposed, float scale)
{
    __shared__ _Float16 As[64][PK_STRIDE];
    __shared__ _Float16 Bs[64][PK_STRIDE];

    const int tid  = threadIdx.x;
    const int w    = tid >> 5;
    const int lane = tid & 31;
    const int m0   = blockIdx.x * 64;
    const int n0   = blockIdx.y * 64;

    const v8f vzero = {0.f,0.f,0.f,0.f,0.f,0.f,0.f,0.f};
    v8f acc[4];
#pragma unroll
    for (int nt = 0; nt < 4; ++nt) acc[nt] = vzero;

    for (int ks = 0; ks < DIMD / 32; ++ks) {
        const int k0 = ks * 32;
        __syncthreads();
        // stage 64x32 fp32 tiles of X and W, converting to f16 in LDS
#pragma unroll
        for (int i = 0; i < 4; ++i) {
            const int chunk = tid + i * 128;     // 0..511
            const int row = chunk >> 3;
            const int c4  = chunk & 7;
            float4 fa = *(const float4*)(X + (size_t)(m0 + row) * DIMD + k0 + c4 * 4);
            v4h ha = { (_Float16)fa.x, (_Float16)fa.y, (_Float16)fa.z, (_Float16)fa.w };
            *(v4h*)&As[row][c4 * 4] = ha;
            float4 fb = *(const float4*)(W + (size_t)(n0 + row) * DIMD + k0 + c4 * 4);
            v4h hb = { (_Float16)fb.x, (_Float16)fb.y, (_Float16)fb.z, (_Float16)fb.w };
            *(v4h*)&Bs[row][c4 * 4] = hb;
        }
        __syncthreads();

        const v16h a = lds_frag(&As[w * 16][0], PK_STRIDE, lane);
#pragma unroll
        for (int nt = 0; nt < 4; ++nt) {
            const v16h b = lds_frag(&Bs[nt * 16][0], PK_STRIDE, lane);
            acc[nt] = wmma16(a, b, acc[nt]);
        }
    }

    // epilogue: C layout -> lane g*16+ln holds rows 8g+r, col ln of each tile
    const int g  = (lane >> 4) & 1;
    const int ln = lane & 15;
#pragma unroll
    for (int nt = 0; nt < 4; ++nt) {
        const int gn = n0 + nt * 16 + ln;
        const int h  = gn >> 6;
        const int hd = gn & 63;
        const float bv = bias[gn];
#pragma unroll
        for (int r = 0; r < 8; ++r) {
            const int gm = m0 + w * 16 + 8 * g + r;
            const int b  = gm >> 12;            // / SEQ
            const int t  = gm & (SEQ - 1);
            const float v = (acc[nt][r] + bv) * scale;
            size_t idx;
            if (vTransposed) idx = ((size_t)(b * NHEADS + h) * HDIM + hd) * SEQ + t;
            else             idx = ((size_t)(b * NHEADS + h) * SEQ + t) * HDIM + hd;
            out[idx] = (_Float16)v;
        }
    }
}

// ---------------------------------------------------------------------------
// FlashAttention core. Grid: (B*H, T/64). 128 threads = 4 waves; wave w owns
// 16 query rows. K/V tiles stream in via TDM (double buffered, TENSORcnt);
// online softmax is wave-local. Q pre-scaled by 1/sqrt(Hd).
// ---------------------------------------------------------------------------
#define AT_STRIDE 72   // 64 cols + 16B pad = 144B per row (matches TDM pad)
#define NSTEP (SEQ / 64)

__global__ void __launch_bounds__(128)
attn_kernel(const _Float16* __restrict__ Qh, const _Float16* __restrict__ Kh,
            const _Float16* __restrict__ Vth, _Float16* __restrict__ Ao)
{
    __shared__ _Float16 Qs[64][AT_STRIDE];
    __shared__ _Float16 Ks[2][64][AT_STRIDE];       // rows = s, cols = d
    __shared__ _Float16 Vs[2][64][AT_STRIDE];       // rows = hd, cols = s
    __shared__ _Float16 Ps[4][16][AT_STRIDE];       // per-wave P staging

    const int tid  = threadIdx.x;
    const int w    = tid >> 5;
    const int lane = tid & 31;
    const int g    = (lane >> 4) & 1;
    const int ln   = lane & 15;
    const int bh   = blockIdx.x;                    // 0..15
    const int t0   = blockIdx.y * 64;

    const size_t qkBase = (size_t)bh * SEQ * HDIM;
    const size_t vBase  = (size_t)bh * HDIM * SEQ;

    // stage 64x64 Q tile once (vector copies)
#pragma unroll
    for (int i = 0; i < 4; ++i) {
        const int chunk = tid + i * 128;            // 0..511
        const int row = chunk >> 3;
        const int c8  = chunk & 7;
        v8h d = *(const v8h*)(Qh + qkBase + (size_t)(t0 + row) * HDIM + c8 * 8);
        *(v8h*)&Qs[row][c8 * 8] = d;
    }
#if HAVE_TDM
    if (w == 0) {   // kick off first K/V tiles; TDM ignores EXEC, one issue/wave
        tdm_load_2d((unsigned)(size_t)&Ks[0][0][0], Kh + qkBase, HDIM);
        tdm_load_2d((unsigned)(size_t)&Vs[0][0][0], Vth + vBase, SEQ);
    }
#endif
    __syncthreads();

    const v16h aq0 = lds_frag(&Qs[w * 16][0],  AT_STRIDE, lane);
    const v16h aq1 = lds_frag(&Qs[w * 16][32], AT_STRIDE, lane);

    const v8f vzero = {0.f,0.f,0.f,0.f,0.f,0.f,0.f,0.f};
    v8f o[4];
#pragma unroll
    for (int nt = 0; nt < 4; ++nt) o[nt] = vzero;
    float mrow[8], lrow[8];
#pragma unroll
    for (int r = 0; r < 8; ++r) { mrow[r] = -3.0e38f; lrow[r] = 0.f; }

    for (int step = 0; step < NSTEP; ++step) {
        const int cur = step & 1;
#if HAVE_TDM
        if (w == 0) {
            if (step + 1 < NSTEP) {      // prefetch next tiles into other buffer
                const int s1 = (step + 1) * 64;
                const int nb = (step + 1) & 1;
                tdm_load_2d((unsigned)(size_t)&Ks[nb][0][0],
                            Kh + qkBase + (size_t)s1 * HDIM, HDIM);
                tdm_load_2d((unsigned)(size_t)&Vs[nb][0][0],
                            Vth + vBase + s1, SEQ);
                WAIT_TCNT(2);            // current tile's 2 loads complete
            } else {
                WAIT_TCNT(0);
            }
        }
#else
        {
            const int s0 = step * 64;
#pragma unroll
            for (int i = 0; i < 4; ++i) {
                const int chunk = tid + i * 128;
                const int row = chunk >> 3;
                const int c8  = chunk & 7;
                v8h dk = *(const v8h*)(Kh + qkBase + (size_t)(s0 + row) * HDIM + c8 * 8);
                *(v8h*)&Ks[cur][row][c8 * 8] = dk;
                v8h dv = *(const v8h*)(Vth + vBase + (size_t)row * SEQ + s0 + c8 * 8);
                *(v8h*)&Vs[cur][row][c8 * 8] = dv;
            }
        }
#endif
        __syncthreads();   // tiles visible to all waves

        // S = Q K^T  (16 x 64 per wave)
        v8f sc[4];
#pragma unroll
        for (int nt = 0; nt < 4; ++nt) {
            const v16h bk0 = lds_frag(&Ks[cur][nt * 16][0],  AT_STRIDE, lane);
            const v16h bk1 = lds_frag(&Ks[cur][nt * 16][32], AT_STRIDE, lane);
            sc[nt] = wmma16(aq0, bk0, vzero);
            sc[nt] = wmma16(aq1, bk1, sc[nt]);
        }

        // online softmax over the 64 new columns (wave-local butterflies)
#pragma unroll
        for (int r = 0; r < 8; ++r) {
            float mx = fmaxf(fmaxf(sc[0][r], sc[1][r]), fmaxf(sc[2][r], sc[3][r]));
            mx = fmaxf(mx, __shfl_xor(mx, 1, 32));
            mx = fmaxf(mx, __shfl_xor(mx, 2, 32));
            mx = fmaxf(mx, __shfl_xor(mx, 4, 32));
            mx = fmaxf(mx, __shfl_xor(mx, 8, 32));
            const float mnew  = fmaxf(mrow[r], mx);
            const float alpha = __expf(mrow[r] - mnew);
            mrow[r] = mnew;
            float rs = 0.f;
#pragma unroll
            for (int nt = 0; nt < 4; ++nt) {
                const float p = __expf(sc[nt][r] - mnew);
                rs += p;
                Ps[w][8 * g + r][nt * 16 + ln] = (_Float16)p;
            }
            rs += __shfl_xor(rs, 1, 32);
            rs += __shfl_xor(rs, 2, 32);
            rs += __shfl_xor(rs, 4, 32);
            rs += __shfl_xor(rs, 8, 32);
            lrow[r] = lrow[r] * alpha + rs;
#pragma unroll
            for (int nt = 0; nt < 4; ++nt) o[nt][r] *= alpha;
        }
        asm volatile("" ::: "memory");   // order P stores before fragment reloads

        // O += P V   (P reloaded in A-fragment layout; wave-private region)
        const v16h ap0 = lds_frag(&Ps[w][0][0],  AT_STRIDE, lane);
        const v16h ap1 = lds_frag(&Ps[w][0][32], AT_STRIDE, lane);
#pragma unroll
        for (int nt = 0; nt < 4; ++nt) {
            const v16h bv0 = lds_frag(&Vs[cur][nt * 16][0],  AT_STRIDE, lane);
            const v16h bv1 = lds_frag(&Vs[cur][nt * 16][32], AT_STRIDE, lane);
            o[nt] = wmma16(ap0, bv0, o[nt]);
            o[nt] = wmma16(ap1, bv1, o[nt]);
        }
        __syncthreads();   // all waves done with this buffer before DMA reuse
    }

    // normalize and write f16 to [B,T,D] (heads re-interleaved)
    const int b = bh / NHEADS, h = bh % NHEADS;
#pragma unroll
    for (int r = 0; r < 8; ++r) {
        const int t = t0 + w * 16 + 8 * g + r;
        const float inv = 1.0f / lrow[r];
#pragma unroll
        for (int nt = 0; nt < 4; ++nt) {
            const int hd = nt * 16 + ln;
            Ao[(size_t)(b * SEQ + t) * DIMD + h * HDIM + hd] = (_Float16)(o[nt][r] * inv);
        }
    }
}

// ---------------------------------------------------------------------------
// Output projection: d_out[m][n] = Ain[m,:] . Wo[n,:] + bo[n]  (fp32 out)
// ---------------------------------------------------------------------------
__global__ void __launch_bounds__(128)
out_proj_kernel(const _Float16* __restrict__ Ain, const float* __restrict__ W,
                const float* __restrict__ bias, float* __restrict__ out)
{
    __shared__ _Float16 As[64][PK_STRIDE];
    __shared__ _Float16 Bs[64][PK_STRIDE];

    const int tid  = threadIdx.x;
    const int w    = tid >> 5;
    const int lane = tid & 31;
    const int m0   = blockIdx.x * 64;
    const int n0   = blockIdx.y * 64;

    const v8f vzero = {0.f,0.f,0.f,0.f,0.f,0.f,0.f,0.f};
    v8f acc[4];
#pragma unroll
    for (int nt = 0; nt < 4; ++nt) acc[nt] = vzero;

    for (int ks = 0; ks < DIMD / 32; ++ks) {
        const int k0 = ks * 32;
        __syncthreads();
        // A tile: already f16 -> straight 16B copies (256 chunks)
#pragma unroll
        for (int i = 0; i < 2; ++i) {
            const int chunk = tid + i * 128;
            const int row = chunk >> 2;
            const int c8  = chunk & 3;
            v8h d = *(const v8h*)(Ain + (size_t)(m0 + row) * DIMD + k0 + c8 * 8);
            *(v8h*)&As[row][c8 * 8] = d;
        }
        // B tile: fp32 -> f16 convert (512 float4 chunks)
#pragma unroll
        for (int i = 0; i < 4; ++i) {
            const int chunk = tid + i * 128;
            const int row = chunk >> 3;
            const int c4  = chunk & 7;
            float4 fb = *(const float4*)(W + (size_t)(n0 + row) * DIMD + k0 + c4 * 4);
            v4h hb = { (_Float16)fb.x, (_Float16)fb.y, (_Float16)fb.z, (_Float16)fb.w };
            *(v4h*)&Bs[row][c4 * 4] = hb;
        }
        __syncthreads();

        const v16h a = lds_frag(&As[w * 16][0], PK_STRIDE, lane);
#pragma unroll
        for (int nt = 0; nt < 4; ++nt) {
            const v16h b = lds_frag(&Bs[nt * 16][0], PK_STRIDE, lane);
            acc[nt] = wmma16(a, b, acc[nt]);
        }
    }

    const int g  = (lane >> 4) & 1;
    const int ln = lane & 15;
#pragma unroll
    for (int nt = 0; nt < 4; ++nt) {
        const int gn = n0 + nt * 16 + ln;
        const float bv = bias[gn];
#pragma unroll
        for (int r = 0; r < 8; ++r) {
            const int gm = m0 + w * 16 + 8 * g + r;
            out[(size_t)gm * DIMD + gn] = acc[nt][r] + bv;
        }
    }
}

// ---------------------------------------------------------------------------
extern "C" void kernel_launch(void* const* d_in, const int* in_sizes, int n_in,
                              void* d_out, int out_size, void* d_ws, size_t ws_size,
                              hipStream_t stream)
{
    (void)in_sizes; (void)n_in; (void)out_size; (void)ws_size;
    const float* query = (const float*)d_in[0];
    const float* key_  = (const float*)d_in[1];
    const float* value = (const float*)d_in[2];
    const float* Wq    = (const float*)d_in[3];
    const float* bq    = (const float*)d_in[4];
    const float* Wk    = (const float*)d_in[5];
    const float* bk    = (const float*)d_in[6];
    const float* Wv    = (const float*)d_in[7];
    const float* bv    = (const float*)d_in[8];
    const float* Wo    = (const float*)d_in[9];
    const float* bo    = (const float*)d_in[10];

    const size_t perBuf = (size_t)MROWS * DIMD;     // 4,194,304 f16 elements
    _Float16* Qh  = (_Float16*)d_ws;                // [B,H,T,Hd] f16 (pre-scaled)
    _Float16* Kh  = Qh  + perBuf;                   // [B,H,T,Hd] f16
    _Float16* Vth = Kh  + perBuf;                   // [B,H,Hd,T] f16 (transposed)
    _Float16* Ao  = Vth + perBuf;                   // [B,T,D]    f16 attention out

    const dim3 gProj(MROWS / 64, DIMD / 64);
    const dim3 bProj(128);

    qkv_proj_kernel<<<gProj, bProj, 0, stream>>>(query, Wq, bq, Qh, 0, 0.125f);
    qkv_proj_kernel<<<gProj, bProj, 0, stream>>>(key_,  Wk, bk, Kh, 0, 1.0f);
    qkv_proj_kernel<<<gProj, bProj, 0, stream>>>(value, Wv, bv, Vth, 1, 1.0f);

    attn_kernel<<<dim3(BATCH * NHEADS, SEQ / 64), dim3(128), 0, stream>>>(Qh, Kh, Vth, Ao);

    out_proj_kernel<<<gProj, bProj, 0, stream>>>(Ao, Wo, bo, (float*)d_out);
}
